// PatchCore_84000970375413
// MI455X (gfx1250) — matrix operand order repaired and checked by reading, not compiled, not run on hardware
//
#include <hip/hip_runtime.h>
#include <hip/hip_bf16.h>

typedef __attribute__((ext_vector_type(16))) __bf16 v16bf;
typedef __attribute__((ext_vector_type(8)))  float  v8f;
typedef __attribute__((ext_vector_type(4)))  int    v4i;

#define NPATCH 784
#define NPAD   896          // 7 * 128
#define DIMK   1536
#define MBANK  156800       // 1225 * 128
#define NT     128
#define MT     128
#define KT     32
#define NK     (DIMK / KT)  // 48
#define A_BYTES (NPAD * DIMK * 2)   // 2,752,512

#if defined(__has_builtin)
# if __has_builtin(__builtin_amdgcn_global_load_async_to_lds_b128) && \
     __has_builtin(__builtin_amdgcn_s_wait_asynccnt)
#  define USE_ASYNC_LDS 1
# endif
# if __has_builtin(__builtin_amdgcn_cvt_pk_bf16_f32)
#  define USE_CVT_PK 1
# endif
# if __has_builtin(__builtin_amdgcn_perm)
#  define USE_PERM_PK 1
# endif
# if __has_builtin(__builtin_amdgcn_sched_barrier)
#  define USE_SCHED_BARRIER 1
# endif
#endif

union FragU { uint4 q[2]; v16bf v; };

#if USE_ASYNC_LDS
// global -> LDS async 16-byte copy (per-lane), tracked by ASYNCcnt
__device__ __forceinline__ void async_ld16(const void* g, void* l) {
    __builtin_amdgcn_global_load_async_to_lds_b128(
        (__attribute__((address_space(1))) v4i*)(void*)(const_cast<void*>(g)),
        (__attribute__((address_space(3))) v4i*)l, 0, 0);
}
#endif

__device__ __forceinline__ void sched_fence() {
#if USE_SCHED_BARRIER
    __builtin_amdgcn_sched_barrier(0);   // pin load issue before the WMMA block
#endif
}

__device__ __forceinline__ unsigned short f2bf(float x) {
    unsigned u = __float_as_uint(x);
    unsigned r = ((u >> 16) & 1u) + 0x7FFFu;   // round-to-nearest-even
    return (unsigned short)((u + r) >> 16);
}
__device__ __forceinline__ unsigned pk_bf16(float x, float y) {
#if USE_CVT_PK
    auto v = __builtin_amdgcn_cvt_pk_bf16_f32(x, y);   // v_cvt_pk_bf16_f32
    unsigned u; __builtin_memcpy(&u, &v, 4);
    return u;
#elif USE_PERM_PK
    // round-bias then merge high halves in one v_perm_b32: {y_hi16, x_hi16}
    unsigned xb = __float_as_uint(x) + 0x8000u;
    unsigned yb = __float_as_uint(y) + 0x8000u;
    return __builtin_amdgcn_perm(yb, xb, 0x07060302u);
#else
    return (unsigned)f2bf(x) | ((unsigned)f2bf(y) << 16);
#endif
}
__device__ __forceinline__ unsigned fenc(float f) {   // order-preserving float->uint
    unsigned u = __float_as_uint(f);
    return (u & 0x80000000u) ? ~u : (u | 0x80000000u);
}
__device__ __forceinline__ float fdec(unsigned u) {
    return __uint_as_float((u & 0x80000000u) ? (u & 0x7FFFFFFFu) : ~u);
}

// ---- combined feature: feat2 channel or 2x bilinear-upsampled feat3 channel ----
__device__ __forceinline__ float combined_val(const float* __restrict__ f2,
                                              const float* __restrict__ f3,
                                              int c, int yy, int xx) {
    if (c < 512) return f2[(c * 28 + yy) * 28 + xx];
    int cc = c - 512;
    float fy = 0.5f * yy - 0.25f, fx = 0.5f * xx - 0.25f;   // half-pixel centers
    int y0 = (int)floorf(fy), x0 = (int)floorf(fx);
    float wy = fy - (float)y0, wx = fx - (float)x0;
    int y0c = min(max(y0, 0), 13), y1c = min(max(y0 + 1, 0), 13);
    int x0c = min(max(x0, 0), 13), x1c = min(max(x0 + 1, 0), 13);
    const float* p = f3 + cc * 196;
    float v00 = p[y0c * 14 + x0c], v01 = p[y0c * 14 + x1c];
    float v10 = p[y1c * 14 + x0c], v11 = p[y1c * 14 + x1c];
    return (1.f - wy) * ((1.f - wx) * v00 + wx * v01) +
           wy        * ((1.f - wx) * v10 + wx * v11);
}

// ---- kernel 1: build normalized bf16 patch matrix A[896][1536], init smax ----
__global__ __launch_bounds__(256)
void patchcore_preprocess(const float* __restrict__ feat2,
                          const float* __restrict__ feat3,
                          unsigned short* __restrict__ A,
                          unsigned* __restrict__ smax) {
    int n = blockIdx.x;           // 0..895
    int tid = threadIdx.x;        // 256 threads
    if (tid == 0) smax[n] = 0u;   // encoded -inf
    if (n >= NPATCH) {            // zero padding rows
        #pragma unroll
        for (int i = 0; i < 6; ++i) A[(size_t)n * DIMK + tid + i * 256] = 0;
        return;
    }
    int y = n / 28, x = n % 28;
    float v[6];
    float ss = 0.f;
    #pragma unroll
    for (int i = 0; i < 6; ++i) {
        int c = tid + i * 256;
        float s = 0.f;
        #pragma unroll
        for (int dy = -1; dy <= 1; ++dy) {
            int yy = y + dy; if (yy < 0 || yy >= 28) continue;
            #pragma unroll
            for (int dx = -1; dx <= 1; ++dx) {
                int xx = x + dx; if (xx < 0 || xx >= 28) continue;
                s += combined_val(feat2, feat3, c, yy, xx);
            }
        }
        v[i] = s * (1.f / 9.f);
        ss += v[i] * v[i];
    }
    __shared__ float red[256];
    red[tid] = ss;
    __syncthreads();
    for (int sft = 128; sft > 0; sft >>= 1) {
        if (tid < sft) red[tid] += red[tid + sft];
        __syncthreads();
    }
    float inv = 1.0f / fmaxf(sqrtf(red[0]), 1e-12f);
    #pragma unroll
    for (int i = 0; i < 6; ++i)
        A[(size_t)n * DIMK + tid + i * 256] = f2bf(v[i] * inv);
}

// ---- kernel 2: fused GEMM (bf16 WMMA) + running row-max over memory bank ----
__global__ __launch_bounds__(256)
void patchcore_gemm_max(const unsigned short* __restrict__ A,
                        const float* __restrict__ bank,
                        unsigned* __restrict__ smax) {
    int nt = blockIdx.x;                 // 0..6   (fast dim -> B-tile reuse in L2)
    int mt = blockIdx.y;                 // 0..1224
    int n0 = nt * NT, m0 = mt * MT;
    int tid = threadIdx.x;
    int lane = tid & 31, wv = tid >> 5;  // 8 waves
    int half = lane >> 4, l16 = lane & 15;

    __shared__ unsigned short As[2][NT][40];   // double-buffered, 16B-aligned chunks
    __shared__ unsigned short Bs[2][MT][40];

    // fixed per-thread staging coordinates
    int ar[2], ac[2];
    #pragma unroll
    for (int i = 0; i < 2; ++i) { int e = (tid * 2 + i) * 8; ar[i] = e >> 5; ac[i] = e & 31; }
    int br[4], bc[4];
    #pragma unroll
    for (int i = 0; i < 4; ++i) { int e = (tid * 4 + i) * 4; br[i] = e >> 5; bc[i] = e & 31; }

    const unsigned short* Ag = A + (size_t)n0 * DIMK;
    const float*          Bg = bank + (size_t)m0 * DIMK;

    v8f acc[8];
    #pragma unroll
    for (int t = 0; t < 8; ++t)
        #pragma unroll
        for (int i = 0; i < 8; ++i) acc[t][i] = 0.f;

    float4 fB[4];
#if !USE_ASYNC_LDS
    uint4 qA[2];
#endif

    // ---- prologue: stage tile 0 into buffer 0 ----
    #pragma unroll
    for (int i = 0; i < 4; ++i)
        fB[i] = *(const float4*)(Bg + (size_t)br[i] * DIMK + bc[i]);
    #pragma unroll
    for (int i = 0; i < 2; ++i) {
#if USE_ASYNC_LDS
        async_ld16(Ag + (size_t)ar[i] * DIMK + ac[i], &As[0][ar[i]][ac[i]]);
#else
        qA[i] = *(const uint4*)(Ag + (size_t)ar[i] * DIMK + ac[i]);
#endif
    }
    #pragma unroll
    for (int i = 0; i < 4; ++i) {
        uint2 p; p.x = pk_bf16(fB[i].x, fB[i].y); p.y = pk_bf16(fB[i].z, fB[i].w);
        *(uint2*)&Bs[0][br[i]][bc[i]] = p;
    }
#if !USE_ASYNC_LDS
    #pragma unroll
    for (int i = 0; i < 2; ++i) *(uint4*)&As[0][ar[i]][ac[i]] = qA[i];
#endif
#if USE_ASYNC_LDS
    __builtin_amdgcn_s_wait_asynccnt(0);
#endif
    __syncthreads();

    // ---- main loop: software-pipelined, double-buffered ----
    for (int kt = 0; kt < NK; ++kt) {
        int cur = kt & 1, nxt = cur ^ 1;
        int k1 = (kt + 1) * KT;
        bool more = (kt + 1 < NK);

        if (more) {   // issue global loads for next tile early
            #pragma unroll
            for (int i = 0; i < 4; ++i)
                fB[i] = *(const float4*)(Bg + (size_t)br[i] * DIMK + k1 + bc[i]);
            #pragma unroll
            for (int i = 0; i < 2; ++i) {
#if USE_ASYNC_LDS
                async_ld16(Ag + (size_t)ar[i] * DIMK + k1 + ac[i], &As[nxt][ar[i]][ac[i]]);
#else
                qA[i] = *(const uint4*)(Ag + (size_t)ar[i] * DIMK + k1 + ac[i]);
#endif
            }
            sched_fence();   // keep the loads issued BEFORE the WMMA block
        }

        // compute on current buffer: fragments + 8 WMMAs
        int arow = wv * 16 + l16;
        FragU ua;
        ua.q[0] = *(const uint4*)&As[cur][arow][half * 8];
        ua.q[1] = *(const uint4*)&As[cur][arow][16 + half * 8];
        FragU ub[8];
        #pragma unroll
        for (int t = 0; t < 8; ++t) {
            int brow = t * 16 + l16;
            ub[t].q[0] = *(const uint4*)&Bs[cur][brow][half * 8];
            ub[t].q[1] = *(const uint4*)&Bs[cur][brow][16 + half * 8];
        }
        #pragma unroll
        for (int t = 0; t < 8; ++t)
            acc[t] = __builtin_amdgcn_wmma_f32_16x16x32_bf16(
                false, ua.v, false, ub[t].v, (short)0, acc[t], false, false);

        if (more) {   // convert + store next tile while WMMAs drain
            #pragma unroll
            for (int i = 0; i < 4; ++i) {
                uint2 p; p.x = pk_bf16(fB[i].x, fB[i].y); p.y = pk_bf16(fB[i].z, fB[i].w);
                *(uint2*)&Bs[nxt][br[i]][bc[i]] = p;
            }
#if !USE_ASYNC_LDS
            #pragma unroll
            for (int i = 0; i < 2; ++i) *(uint4*)&As[nxt][ar[i]][ac[i]] = qA[i];
#endif
        }
#if USE_ASYNC_LDS
        __builtin_amdgcn_s_wait_asynccnt(0);
#endif
        __syncthreads();
    }

    // per-lane max over the 8 m-chunks, then butterfly over the 16-lane half
    float mx[8];
    #pragma unroll
    for (int v = 0; v < 8; ++v) {
        float m = acc[0][v];
        #pragma unroll
        for (int t = 1; t < 8; ++t) m = fmaxf(m, acc[t][v]);
        mx[v] = m;
    }
    #pragma unroll
    for (int d = 1; d < 16; d <<= 1)
        #pragma unroll
        for (int v = 0; v < 8; ++v)
            mx[v] = fmaxf(mx[v], __shfl_xor(mx[v], d, 32));

    if (l16 == 0) {
        #pragma unroll
        for (int v = 0; v < 8; ++v) {
            int n = n0 + wv * 16 + v + 8 * half;   // C layout: M = v (+8 for hi half)
            if (n < NPATCH) atomicMax(&smax[n], fenc(mx[v]));
        }
    }
}

// ---- kernel 3: dists, score, 28->224 bilinear heatmap, min-max normalize ----
__global__ __launch_bounds__(256)
void patchcore_finalize(const unsigned* __restrict__ smax,
                        float* __restrict__ out) {
    __shared__ float nn[NPATCH];
    __shared__ float red[256];
    __shared__ float red2[256];
    __shared__ float s_min, s_max, s_score;
    int tid = threadIdx.x;

    float lsc = -1e30f;
    for (int n = tid; n < NPATCH; n += 256) {
        float s = fdec(smax[n]);
        float d2 = 2.f - 2.f * s;                 // ||p||=||m||=1
        float d = sqrtf(fmaxf(d2, 0.f));
        nn[n] = d;
        lsc = fmaxf(lsc, d);
    }
    red[tid] = lsc;
    __syncthreads();
    for (int sft = 128; sft > 0; sft >>= 1) {
        if (tid < sft) red[tid] = fmaxf(red[tid], red[tid + sft]);
        __syncthreads();
    }
    if (tid == 0) s_score = red[0];
    __syncthreads();

    // bilinear upsample 28x28 -> 224x224, half-pixel centers, clamped edges
    float lmin = 1e30f, lmax = -1e30f;
    for (int idx = tid; idx < 224 * 224; idx += 256) {
        int oy = idx / 224, ox = idx % 224;
        float fy = 0.125f * oy - 0.4375f;
        float fx = 0.125f * ox - 0.4375f;
        int y0 = (int)floorf(fy), x0 = (int)floorf(fx);
        float wy = fy - (float)y0, wx = fx - (float)x0;
        int y0c = min(max(y0, 0), 27), y1c = min(max(y0 + 1, 0), 27);
        int x0c = min(max(x0, 0), 27), x1c = min(max(x0 + 1, 0), 27);
        float v00 = nn[y0c * 28 + x0c], v01 = nn[y0c * 28 + x1c];
        float v10 = nn[y1c * 28 + x0c], v11 = nn[y1c * 28 + x1c];
        float h = (1.f - wy) * ((1.f - wx) * v00 + wx * v01) +
                  wy        * ((1.f - wx) * v10 + wx * v11);
        out[1 + idx] = h;
        lmin = fminf(lmin, h);
        lmax = fmaxf(lmax, h);
    }
    red[tid] = lmin;
    red2[tid] = lmax;
    __syncthreads();
    for (int sft = 128; sft > 0; sft >>= 1) {
        if (tid < sft) {
            red[tid]  = fminf(red[tid],  red[tid + sft]);
            red2[tid] = fmaxf(red2[tid], red2[tid + sft]);
        }
        __syncthreads();
    }
    if (tid == 0) { s_min = red[0]; s_max = red2[0]; }
    __syncthreads();

    float hmin = s_min, hmax = s_max;
    if (hmax > hmin) {
        float inv = 1.0f / (hmax - hmin);
        for (int idx = tid; idx < 224 * 224; idx += 256)
            out[1 + idx] = (out[1 + idx] - hmin) * inv;
    }
    if (tid == 0) out[0] = s_score;
}

extern "C" void kernel_launch(void* const* d_in, const int* in_sizes, int n_in,
                              void* d_out, int out_size, void* d_ws, size_t ws_size,
                              hipStream_t stream) {
    const float* feat2 = (const float*)d_in[0];   // [512,28,28]
    const float* feat3 = (const float*)d_in[1];   // [1024,14,14]
    const float* bank  = (const float*)d_in[2];   // [156800,1536]
    float* out = (float*)d_out;                   // [1 + 224*224]

    unsigned short* A = (unsigned short*)d_ws;                 // 896x1536 bf16
    unsigned* smax = (unsigned*)((char*)d_ws + A_BYTES);       // 896 encoded maxima

    patchcore_preprocess<<<NPAD, 256, 0, stream>>>(feat2, feat3, A, smax);

    dim3 grid(NPAD / NT, MBANK / MT);   // (7, 1225): n fast -> bank tile L2 reuse
    patchcore_gemm_max<<<grid, 256, 0, stream>>>(A, bank, smax);

    patchcore_finalize<<<1, 256, 0, stream>>>(smax, out);
}
